// QCNN_61890478736067
// MI455X (gfx1250) — matrix operand order repaired and checked by reading, not compiled, not run on hardware
//
#include <hip/hip_runtime.h>
#include <hip/hip_bf16.h>

typedef _Float16 v16h __attribute__((ext_vector_type(16)));
typedef _Float16 v8h  __attribute__((ext_vector_type(8)));
typedef float    v8f  __attribute__((ext_vector_type(8)));

struct U2 { float r00, i00, r01, i01, r10, i10, r11, i11; };

// Apply a 2x2 complex unitary to qubit with flat-index bit `mask` (8>>wire).
__device__ __forceinline__ void apply1q(float sr[16], float si[16], const U2& u, int mask) {
#pragma unroll
    for (int i = 0; i < 16; ++i) {
        if (i & mask) continue;
        const int j = i | mask;
        float ar = sr[i], ai = si[i], br = sr[j], bi = si[j];
        sr[i] = u.r00 * ar - u.i00 * ai + u.r01 * br - u.i01 * bi;
        si[i] = u.r00 * ai + u.i00 * ar + u.r01 * bi + u.i01 * br;
        sr[j] = u.r10 * ar - u.i10 * ai + u.r11 * br - u.i11 * bi;
        si[j] = u.r10 * ai + u.i10 * ar + u.r11 * bi + u.i11 * br;
    }
}

// CNOT: control mask mc, target mask mt (mask = 8>>wire). Pure amplitude swap.
__device__ __forceinline__ void cnotg(float sr[16], float si[16], int mc, int mt) {
#pragma unroll
    for (int i = 0; i < 16; ++i) {
        if ((i & mc) && !(i & mt)) {
            const int j = i | mt;
            float tr = sr[i]; sr[i] = sr[j]; sr[j] = tr;
            float ti = si[i]; si[i] = si[j]; si[j] = ti;
        }
    }
}

// One scan step: Rot(U) on each wire, each followed by the CNOT ring.
__device__ __forceinline__ void stepfn(float sr[16], float si[16], const U2& u) {
#pragma unroll
    for (int w = 0; w < 4; ++w) {
        apply1q(sr, si, u, 8 >> w);
        cnotg(sr, si, 8, 4);   // cnot(0,1)
        cnotg(sr, si, 4, 2);   // cnot(1,2)
        cnotg(sr, si, 2, 1);   // cnot(2,3)
        cnotg(sr, si, 1, 8);   // cnot(3,0)
    }
}

// One thread per patch: full 4-qubit statevector sim in VGPRs.
// Writes 4 PauliZ expvals as f16 into qbuf (row-major 64 x 800, K-padded).
__global__ void __launch_bounds__(256) sim_kernel(const float* __restrict__ images,
                                                  const float* __restrict__ rp,
                                                  _Float16* __restrict__ qbuf) {
    const int tid = blockIdx.x * 256 + threadIdx.x;   // exactly 64*196 = 12544 threads
    const int b = tid / 196;
    const int p = tid - b * 196;
    const int pj = p / 14, pk = p - pj * 14;
    const float* img = images + b * 784;
    float px[4];
    px[0] = img[(2 * pj) * 28 + 2 * pk];
    px[1] = img[(2 * pj) * 28 + 2 * pk + 1];
    px[2] = img[(2 * pj + 1) * 28 + 2 * pk];
    px[3] = img[(2 * pj + 1) * 28 + 2 * pk + 1];

    const float PI = 3.14159265358979323846f;
    // qml.Rot = RZ(phi)->RY(theta)->RZ(omega); only 4 distinct unitaries (GROUP_IDX period 4)
    auto mkrot = [&](int i0, int i1, int i2) {
        float phi = px[i0] * PI, th = px[i1] * PI, om = px[i2] * PI;
        float st, ct; __sincosf(0.5f * th, &st, &ct);
        float a = 0.5f * (phi + om), d = 0.5f * (phi - om);
        float sa, ca, sd, cd;
        __sincosf(a, &sa, &ca);
        __sincosf(d, &sd, &cd);
        U2 u;
        u.r00 =  ca * ct; u.i00 = -sa * ct;   // ep*c
        u.r01 = -cd * st; u.i01 = -sd * st;   // -em*s
        u.r10 =  cd * st; u.i10 = -sd * st;   // conj(em)*s
        u.r11 =  ca * ct; u.i11 =  sa * ct;   // conj(ep)*c
        return u;
    };
    U2 u0 = mkrot(0, 1, 2), u1 = mkrot(3, 0, 1), u2 = mkrot(2, 3, 0), u3 = mkrot(1, 2, 3);

    float sr[16], si[16];
#pragma unroll
    for (int i = 0; i < 16; ++i) { sr[i] = 0.0f; si[i] = 0.0f; }
    sr[0] = 1.0f;

#pragma unroll 1
    for (int t = 0; t < 5; ++t) {   // 20 steps, U pattern repeats every 4
        stepfn(sr, si, u0);
        stepfn(sr, si, u1);
        stepfn(sr, si, u2);
        stepfn(sr, si, u3);
    }

    // RandomLayers stand-in: rx(p0)@w3, ry(p1)@w1, cnot(0,2), rz(p2)@w0, rx(p3)@w2, cnot(1,3)
    float q0 = rp[0], q1 = rp[1], q2 = rp[2], q3 = rp[3];
    { float s, c; __sincosf(0.5f * q0, &s, &c); U2 u{c, 0, 0, -s, 0, -s, c, 0}; apply1q(sr, si, u, 1); }
    { float s, c; __sincosf(0.5f * q1, &s, &c); U2 u{c, 0, -s, 0, s, 0, c, 0};  apply1q(sr, si, u, 4); }
    cnotg(sr, si, 8, 2);
    { float s, c; __sincosf(0.5f * q2, &s, &c); U2 u{c, -s, 0, 0, 0, 0, c, s};  apply1q(sr, si, u, 8); }
    { float s, c; __sincosf(0.5f * q3, &s, &c); U2 u{c, 0, 0, -s, 0, -s, c, 0}; apply1q(sr, si, u, 2); }
    cnotg(sr, si, 4, 1);

    float ev0 = 0.f, ev1 = 0.f, ev2 = 0.f, ev3 = 0.f;
#pragma unroll
    for (int i = 0; i < 16; ++i) {
        float pr = sr[i] * sr[i] + si[i] * si[i];
        ev0 += (i & 8) ? -pr : pr;
        ev1 += (i & 4) ? -pr : pr;
        ev2 += (i & 2) ? -pr : pr;
        ev3 += (i & 1) ? -pr : pr;
    }
    _Float16* qo = qbuf + b * 800 + p * 4;
    qo[0] = (_Float16)ev0; qo[1] = (_Float16)ev1;
    qo[2] = (_Float16)ev2; qo[3] = (_Float16)ev3;
}

// Stage fc_w as f16 B-fragments (exact v_wmma 16x16x32 B layout, K padded to 800,
// N padded to 16 with zeros) and zero qbuf's K padding.
__global__ void prep_kernel(const float* __restrict__ fc_w,
                            _Float16* __restrict__ wbuf,
                            _Float16* __restrict__ qbuf) {
    const int tid = blockIdx.x * blockDim.x + threadIdx.x;
    if (tid < 12800) {                       // 25 K-tiles * 32 lanes * 16 halfs
        int kt   = tid >> 9;                 // /512
        int r    = tid & 511;
        int lane = r >> 4;
        int t    = r & 15;
        int n    = lane & 15;                // column (class)
        int k    = kt * 32 + ((lane < 16) ? 0 : 8) + ((t < 8) ? t : t + 8);
        float v  = (k < 784 && n < 10) ? fc_w[n * 784 + k] : 0.0f;
        wbuf[tid] = (_Float16)v;
    }
    if (tid < 1024) {                        // zero q K-padding cols 784..799
        int b = tid >> 4;
        int c = tid & 15;
        qbuf[b * 800 + 784 + c] = (_Float16)0.0f;
    }
}

// logits(64x16) = q(64x800,f16) * Wt(800x16,f16) via v_wmma_f32_16x16x32_f16.
// One wave per 16-row M-tile; 25 K-steps of 32.
__global__ void __launch_bounds__(32) fc_wmma_kernel(const _Float16* __restrict__ qbuf,
                                                     const _Float16* __restrict__ wbuf,
                                                     float* __restrict__ logits) {
    const int lane  = threadIdx.x;           // 0..31, wave32, EXEC all ones
    const int mtile = blockIdx.x;            // 0..3
    const int hsel  = lane >> 4;             // 0: K {0-7,16-23}; 1: K {8-15,24-31}
    const int row   = mtile * 16 + (lane & 15);
    v8f acc = {};
#pragma unroll 1
    for (int kt = 0; kt < 25; ++kt) {
        const int k0 = kt * 32;
        // A fragment: two contiguous 16B chunks per lane (ISA 16-bit A 16x32 layout)
        const _Float16* aq = qbuf + row * 800 + k0 + hsel * 8;
        v8h lo = *(const v8h*)(aq);
        v8h hi = *(const v8h*)(aq + 16);
        v16h a = __builtin_shufflevector(lo, hi, 0, 1, 2, 3, 4, 5, 6, 7,
                                         8, 9, 10, 11, 12, 13, 14, 15);
        // B fragment: pre-swizzled, one contiguous 32B load per lane
        v16h bm = *(const v16h*)(wbuf + kt * 512 + lane * 16);
        acc = __builtin_amdgcn_wmma_f32_16x16x32_f16(
            /*neg_a=*/false, a, /*neg_b=*/false, bm,
            /*c_mod=*/(short)0, acc, /*reuse_a=*/false, /*reuse_b=*/false);
    }
    // D layout: VGPR r -> (M = r + 8*hsel, N = lane&15)
    const int col   = lane & 15;
    const int rbase = mtile * 16 + hsel * 8;
#pragma unroll
    for (int r = 0; r < 8; ++r)
        logits[(rbase + r) * 16 + col] = acc[r];
}

// Bias + numerically-stable log_softmax over the 10 valid classes.
__global__ void __launch_bounds__(64) finalize_kernel(const float* __restrict__ logits,
                                                      const float* __restrict__ fc_b,
                                                      float* __restrict__ out) {
    const int row = threadIdx.x;             // 64 rows
    float x[10];
    float m = -1e30f;
#pragma unroll
    for (int c = 0; c < 10; ++c) {
        x[c] = logits[row * 16 + c] + fc_b[c];
        m = fmaxf(m, x[c]);
    }
    float s = 0.0f;
#pragma unroll
    for (int c = 0; c < 10; ++c) s += __expf(x[c] - m);
    float lse = m + __logf(s);
#pragma unroll
    for (int c = 0; c < 10; ++c) out[row * 10 + c] = x[c] - lse;
}

extern "C" void kernel_launch(void* const* d_in, const int* in_sizes, int n_in,
                              void* d_out, int out_size, void* d_ws, size_t ws_size,
                              hipStream_t stream) {
    const float* images      = (const float*)d_in[0];   // (64,1,28,28)
    const float* rand_params = (const float*)d_in[1];   // (1,4)
    const float* fc_w        = (const float*)d_in[2];   // (10,784)
    const float* fc_b        = (const float*)d_in[3];   // (10,)
    float* out = (float*)d_out;                         // (64,10) f32

    char* ws = (char*)d_ws;
    _Float16* qbuf = (_Float16*)(ws);                   // 64*800 f16  = 102400 B
    _Float16* wbuf = (_Float16*)(ws + 102400);          // 25*512 f16 =  25600 B
    float*  logits = (float*)(ws + 128000);             // 64*16 f32  =   4096 B

    sim_kernel<<<49, 256, 0, stream>>>(images, rand_params, qbuf);     // 12544 patches
    prep_kernel<<<(12800 + 255) / 256, 256, 0, stream>>>(fc_w, wbuf, qbuf);
    fc_wmma_kernel<<<4, 32, 0, stream>>>(qbuf, wbuf, logits);
    finalize_kernel<<<1, 64, 0, stream>>>(logits, fc_b, out);
}
